// LCritic_39737037422777
// MI455X (gfx1250) — matrix operand order repaired and checked by reading, not compiled, not run on hardware
//
#include <hip/hip_runtime.h>

// ---------------------------------------------------------------------------
// LSTM-critic forward for MI455X (gfx1250, wave32, WMMA).
//  x = relu([state|action] @ W_in + b_in)            GEMM1: [B,224]x[224,64]
//  gates = x @ W_i + hidden @ W_h + b_i + b_h        GEMM2: 2x [B,64]x[64,256]
//  i,f,g,o -> c = f*cell + i*g ; h = o*tanh(c)
//  out = tanh(h @ W_out + b_out)
// Memory-bound (~252 MB @ 23.3 TB/s ~ 11us); math done as f16 WMMA w/ f32 acc.
// f32->f16 packing uses v_cvt_pk_rtz_f16_f32 (1 VALU op per pair).
// ---------------------------------------------------------------------------

typedef __attribute__((ext_vector_type(16))) _Float16     v16h;
typedef __attribute__((ext_vector_type(2)))  __fp16       v2fp16;
typedef __attribute__((ext_vector_type(8)))  float        v8f;
typedef __attribute__((ext_vector_type(4)))  unsigned int u32x4;

#define BATCH 131072
#define HDIM  64
#define INW   224

// --- LDS layout (bytes). Strides padded: rows 16B aligned + bank-spread. ---
#define WKP   232                         // f16 K-stride for WinT rows (464B)
#define KP2   72                          // f16 K-stride for WiT/WhT/xh/hid rows (144B)
#define XKP   48                          // f16 K-stride for xin chunk rows (96B)
#define OFF_WINT 0
#define SZ_WINT  (64 * WKP * 2)           // 29696
#define OFF_WIT  (OFF_WINT + SZ_WINT)
#define SZ_WIT   (256 * KP2 * 2)          // 36864
#define OFF_WHT  (OFF_WIT + SZ_WIT)
#define OFF_WAVE (OFF_WHT + SZ_WIT)       // 103424
#define WAVE_SZ  6144
#define OFF_XIN  0                        // 16*48*2 = 1536
#define OFF_XH   1536                     // 16*72*2 = 2304
#define OFF_HID  3840                     // 16*72*2 = 2304
#define LDS_TOTAL (OFF_WAVE + 8 * WAVE_SZ)  // 152576 (< 160KB -> 2 WGs/WGP)

// One v_cvt_pk_rtz_f16_f32: two f32 -> packed 2xf16 in a single VALU op.
static __device__ __forceinline__ unsigned pk2(float a, float b) {
    union { v2fp16 h; unsigned u; } t;
    t.h = __builtin_amdgcn_cvt_pkrtz(a, b);
    return t.u;
}

// Two 16B LDS reads -> one 16-element f16 fragment.
static __device__ __forceinline__ v16h frag_ld(const char* p0, const char* p1) {
    union { u32x4 u[2]; v16h h; } t;
    t.u[0] = *(const u32x4*)p0;
    t.u[1] = *(const u32x4*)p1;
    return t.h;
}

static __device__ __forceinline__ float sigmoidf_(float x) {
    return 1.0f / (1.0f + __expf(-x));
}
static __device__ __forceinline__ float tanhf_(float x) {
    return 1.0f - 2.0f / (__expf(2.0f * x) + 1.0f);
}

__global__ void __launch_bounds__(256, 2)
lstm_critic_wmma(const float* __restrict__ state,  const float* __restrict__ action,
                 const float* __restrict__ hidden, const float* __restrict__ cell,
                 const float* __restrict__ W_in,   const float* __restrict__ b_in,
                 const float* __restrict__ W_i,    const float* __restrict__ b_i,
                 const float* __restrict__ W_h,    const float* __restrict__ b_h,
                 const float* __restrict__ W_out,  const float* __restrict__ b_out,
                 float* __restrict__ out) {
    extern __shared__ char smem[];
    const int tid    = threadIdx.x;
    const int lane   = tid & 31;
    const int wave   = tid >> 5;
    const int half16 = lane >> 4;   // GEMM1: k 0-15 | 16-31; GEMM2: k 0-7,16-23 | 8-15,24-31 (A/B)
    const int n16    = lane & 15;   // column (B/C/D frags) / row (A frags)

    // ---- stage weights into LDS as f16, transposed [N][Kpad] ----
    for (int idx = tid; idx < INW * HDIM; idx += 256) {
        int k = idx >> 6, n = idx & 63;
        *(_Float16*)(smem + OFF_WINT + (n * WKP + k) * 2) = (_Float16)W_in[idx];
    }
    for (int idx = tid; idx < HDIM * 4 * HDIM; idx += 256) {
        int k = idx >> 8, n = idx & 255;
        *(_Float16*)(smem + OFF_WIT + (n * KP2 + k) * 2) = (_Float16)W_i[idx];
        *(_Float16*)(smem + OFF_WHT + (n * KP2 + k) * 2) = (_Float16)W_h[idx];
    }
    __syncthreads();

    char* mywave = smem + OFF_WAVE + wave * WAVE_SZ;  // wave-private staging
    char* xin = mywave + OFF_XIN;
    char* xh  = mywave + OFF_XH;
    char* hid = mywave + OFF_HID;

    float wo[4];
#pragma unroll
    for (int t = 0; t < 4; ++t) wo[t] = W_out[t * 16 + n16];
    const float bout_s = b_out[0];

    float* __restrict__ out_h = out + BATCH;
    float* __restrict__ out_c = out + BATCH + (long)BATCH * HDIM;

    const int nTiles = BATCH / 16;  // 8192
    for (int tile = blockIdx.x * 8 + wave; tile < nTiles; tile += (int)gridDim.x * 8) {
        const int r0 = tile * 16;

        // ---- stage hidden tile: 16x64 f32 -> f16 [16][KP2] ----
#pragma unroll
        for (int it = 0; it < 8; ++it) {
            int i4 = it * 32 + lane;          // 0..255 float4s
            int m = i4 >> 4, q = i4 & 15;
            const float4 v = *(const float4*)(hidden + (long)(r0 + m) * HDIM + q * 4);
            unsigned* dst = (unsigned*)(hid + (m * KP2 + q * 4) * 2);
            dst[0] = pk2(v.x, v.y);
            dst[1] = pk2(v.z, v.w);
        }

        // ---- GEMM1: x = relu([state|action] @ W_in + b_in), K streamed 7x32 ----
        v8f cx[4];
#pragma unroll
        for (int t = 0; t < 4; ++t) {
            float bv = b_in[t * 16 + n16];
#pragma unroll
            for (int j = 0; j < 8; ++j) cx[t][j] = bv;
        }
        for (int kc = 0; kc < 7; ++kc) {
            // stage 16x32 f32 chunk -> f16 [16][XKP]
#pragma unroll
            for (int it = 0; it < 4; ++it) {
                int i4 = it * 32 + lane;      // 0..127 float4s
                int m = i4 >> 3, q = i4 & 7;
                const float* src = (kc < 6)
                    ? (state  + (long)(r0 + m) * 192 + kc * 32 + q * 4)
                    : (action + (long)(r0 + m) * 32  + q * 4);
                const float4 v = *(const float4*)src;
                unsigned* dst = (unsigned*)(xin + (m * XKP + q * 4) * 2);
                dst[0] = pk2(v.x, v.y);
                dst[1] = pk2(v.z, v.w);
            }
            const char* arow = xin + n16 * (XKP * 2);
            const v16h a = frag_ld(arow + 32 * half16, arow + 16 + 32 * half16);
#pragma unroll
            for (int t = 0; t < 4; ++t) {
                const char* brow = smem + OFF_WINT + (t * 16 + n16) * (WKP * 2)
                                 + kc * 64 + 32 * half16;
                const v16h b = frag_ld(brow, brow + 16);
                cx[t] = __builtin_amdgcn_wmma_f32_16x16x32_f16(
                    false, a, false, b, (short)0, cx[t], false, false);
            }
        }

        // ---- relu + stash x tile (f16) for GEMM2 A-fragments ----
#pragma unroll
        for (int t = 0; t < 4; ++t) {
#pragma unroll
            for (int j = 0; j < 8; ++j) {
                float v = cx[t][j];
                v = v > 0.0f ? v : 0.0f;
                int m = j + 8 * half16;
                *(_Float16*)(xh + (m * KP2 + t * 16 + n16) * 2) = (_Float16)v;
            }
        }

        // ---- GEMM2 A fragments (x and hidden, K=64 -> 2 chunks each) ----
        const char* xrow = xh  + n16 * (KP2 * 2);
        const char* hrow = hid + n16 * (KP2 * 2);
        const v16h ax0 = frag_ld(xrow + 16 * half16,      xrow + 32 + 16 * half16);
        const v16h ax1 = frag_ld(xrow + 64 + 16 * half16, xrow + 96 + 16 * half16);
        const v16h ah0 = frag_ld(hrow + 16 * half16,      hrow + 32 + 16 * half16);
        const v16h ah1 = frag_ld(hrow + 64 + 16 * half16, hrow + 96 + 16 * half16);

        float s[8];
#pragma unroll
        for (int j = 0; j < 8; ++j) s[j] = 0.0f;

#pragma unroll
        for (int nt = 0; nt < 4; ++nt) {
            const int c = nt * 16 + n16;
            v8f gi, gf, gg, go;
            {
                const float bi_ = b_i[c]       + b_h[c];
                const float bf_ = b_i[64 + c]  + b_h[64 + c];
                const float bg_ = b_i[128 + c] + b_h[128 + c];
                const float bo_ = b_i[192 + c] + b_h[192 + c];
#pragma unroll
                for (int j = 0; j < 8; ++j) { gi[j] = bi_; gf[j] = bf_; gg[j] = bg_; go[j] = bo_; }
            }
#pragma unroll
            for (int kc = 0; kc < 2; ++kc) {
                const v16h ax = kc ? ax1 : ax0;
                const v16h ah = kc ? ah1 : ah0;
                const int kb = kc * 64 + 16 * half16;
#pragma unroll
                for (int g = 0; g < 4; ++g) {       // gate columns: g*64 + nt*16
                    const int rowoff = (g * 64 + nt * 16 + n16) * (KP2 * 2) + kb;
                    const char* bi_p = smem + OFF_WIT + rowoff;
                    const char* bh_p = smem + OFF_WHT + rowoff;
                    const v16h bwi = frag_ld(bi_p, bi_p + 32);
                    const v16h bwh = frag_ld(bh_p, bh_p + 32);
                    v8f* acc = (g == 0) ? &gi : (g == 1) ? &gf : (g == 2) ? &gg : &go;
                    *acc = __builtin_amdgcn_wmma_f32_16x16x32_f16(
                        false, ax, false, bwi, (short)0, *acc, false, false);
                    *acc = __builtin_amdgcn_wmma_f32_16x16x32_f16(
                        false, ah, false, bwh, (short)0, *acc, false, false);
                }
            }
            // ---- elementwise gates + h/c stores (C-fragment layout) ----
#pragma unroll
            for (int j = 0; j < 8; ++j) {
                const float iv = sigmoidf_(gi[j]);
                const float fv = sigmoidf_(gf[j]);
                const float gv = tanhf_(gg[j]);
                const float ov = sigmoidf_(go[j]);
                const int  m   = j + 8 * half16;
                const long idx = (long)(r0 + m) * HDIM + c;
                const float cv = fv * cell[idx] + iv * gv;
                const float hv = ov * tanhf_(cv);
                out_h[idx] = hv;
                out_c[idx] = cv;
                s[j] += hv * wo[nt];
            }
        }

        // ---- output head: reduce h·W_out across the 16-lane column group ----
#pragma unroll
        for (int j = 0; j < 8; ++j) {
            float v = s[j];
            v += __shfl_xor(v, 1, 16);
            v += __shfl_xor(v, 2, 16);
            v += __shfl_xor(v, 4, 16);
            v += __shfl_xor(v, 8, 16);
            s[j] = v;
        }
        if (n16 < 8) {
            float v = s[0];
#pragma unroll
            for (int j = 1; j < 8; ++j) v = (n16 == j) ? s[j] : v;
            out[r0 + n16 + 8 * half16] = tanhf_(v + bout_s);
        }
    }
}

extern "C" void kernel_launch(void* const* d_in, const int* in_sizes, int n_in,
                              void* d_out, int out_size, void* d_ws, size_t ws_size,
                              hipStream_t stream) {
    (void)in_sizes; (void)n_in; (void)out_size; (void)d_ws; (void)ws_size;
    const float* state  = (const float*)d_in[0];
    const float* action = (const float*)d_in[1];
    const float* hidden = (const float*)d_in[2];
    const float* cell   = (const float*)d_in[3];
    const float* W_in   = (const float*)d_in[4];
    const float* b_in   = (const float*)d_in[5];
    const float* W_i    = (const float*)d_in[6];
    const float* b_i    = (const float*)d_in[7];
    const float* W_h    = (const float*)d_in[8];
    const float* b_h    = (const float*)d_in[9];
    const float* W_out  = (const float*)d_in[10];
    const float* b_out  = (const float*)d_in[11];
    float* out = (float*)d_out;

    dim3 grid(512), block(256);
    lstm_critic_wmma<<<grid, block, LDS_TOTAL, stream>>>(
        state, action, hidden, cell, W_in, b_in, W_i, b_i, W_h, b_h, W_out, b_out, out);
}
